// MultiGATConvGRU_80539226734912
// MI455X (gfx1250) — matrix operand order repaired and checked by reading, not compiled, not run on hardware
//
#include <hip/hip_runtime.h>
#include <hip/hip_bf16.h>

// ---------------------------------------------------------------------------
// MultiGATConvGRU for MI455X (gfx1250, wave32, WMMA).
// All dense transforms (MLPs, GAT linear, GRU gates) run through
// v_wmma_f32_16x16x32_f16.  Each wave computes a 16x64 output block
// (4 n-tiles register-blocked: one A fragment feeds 4 WMMAs).  Weights are
// pre-packed into WMMA B-fragment-major layout so each lane's B fragment is
// one contiguous 32B load.  Irregular edge softmax / aggregation uses f32
// atomics with 4-channel vectorization.
// Workspace use: ~400 MB (node activations in f16, f32 atomic agg arena).
// ---------------------------------------------------------------------------

typedef __attribute__((ext_vector_type(16))) _Float16 v16h;
typedef __attribute__((ext_vector_type(8)))  _Float16 v8h;
typedef __attribute__((ext_vector_type(8)))  float    v8f;

#define NN1 100000   // factor nodes  (multiple of 16)
#define NN2 50000    // variable nodes (multiple of 16)
#define NE  300000
#define NT  3
#define NS  64       // NSTATE
#define NH  3        // heads
#define NHC 192      // heads * NSTATE

// ---------------------------------------------------------------------------
// WMMA GEMM: C[M x ncols] = act(A[M x K](f16) * Bpacked[K x Np](f16) + bias)
//   grid = (Np/(16*NB), M/16), block = 32 (one wave per 16 x 16*NB block).
//   A row-major, K multiple of 32.  Bpacked is fragment-major (see k_pack).
//   Stores f16 (Ch) and/or f32 (Cf), column-guarded by ncols.
// ---------------------------------------------------------------------------
template<int NB>
__global__ __launch_bounds__(32) void k_gemm(
    const _Float16* __restrict__ A, const _Float16* __restrict__ Bp,
    const float* __restrict__ bias, _Float16* __restrict__ Ch,
    float* __restrict__ Cf, int K, int ncols, int act)
{
  const int lane = threadIdx.x;
  const int lo = lane & 15, hi = lane >> 4;
  const int kch = K >> 5;
  const long tm = blockIdx.y;
  const long tn0 = (long)blockIdx.x * NB;
  const _Float16* __restrict__ Arow = A + (tm * 16 + lo) * (long)K;
  const _Float16* __restrict__ Bt   = Bp + tn0 * (long)kch * 512 + lane * 16;
  v8f c[NB] = {};
  for (int cc = 0; cc < kch; ++cc) {
    // A fragment: row M=lo; lane-half hi selects K {0..15} / {16..31}
    const _Float16* ap = Arow + cc * 32 + hi * 16;
    v16h a = *(const v16h*)(ap);
    __builtin_prefetch(Arow + (cc + 1) * 32, 0, 1);   // global_prefetch_b8
#pragma unroll
    for (int q = 0; q < NB; ++q) {
      // B fragment: packed contiguously per (tile, chunk, lane)
      v16h b = *(const v16h*)(Bt + ((long)q * kch + cc) * 512);
      c[q] = __builtin_amdgcn_wmma_f32_16x16x32_f16(false, a, false, b,
                                                    (short)0, c[q], false, false);
    }
  }
#pragma unroll
  for (int q = 0; q < NB; ++q) {
    const int n = (int)(tn0 + q) * 16 + lo;
    if (n >= ncols) continue;
    const float bv = bias ? bias[n] : 0.f;
#pragma unroll
    for (int r = 0; r < 8; ++r) {
      long row = tm * 16 + hi * 8 + r;   // C/D layout: VGPR r -> M = r + hi*8
      float v = c[q][r] + bv;
      if (act) v = v > 0.f ? v : 0.f;
      if (Ch) Ch[row * (long)ncols + n] = (_Float16)v;
      if (Cf) Cf[row * (long)ncols + n] = v;
    }
  }
}

// Pack f32 weight [Kr x Nr] (or its transpose) into WMMA B-fragment layout,
// zero-padded to [Kp x Np]:  dst[(((ntile*kch)+cc)*32 + lane)*16 + j]
//   where n = ntile*16 + (lane&15),  k = cc*32 + (lane>>4)*16 + j.
__global__ void k_pack(const float* __restrict__ s, _Float16* __restrict__ d,
                       int Kr, int Nr, int Kp, int Np, int tr)
{
  long i = (long)blockIdx.x * blockDim.x + threadIdx.x;
  long tot = (long)(Np >> 4) * (Kp >> 5) * 512;
  if (i >= tot) return;
  int j = (int)(i & 15);
  int lane = (int)((i >> 4) & 31);
  long rest = i >> 9;
  int kch = Kp >> 5;
  int cc = (int)(rest % kch);
  int ntile = (int)(rest / kch);
  int n = ntile * 16 + (lane & 15);
  int k = cc * 32 + (lane >> 4) * 16 + j;
  float v = 0.f;
  if (k < Kr && n < Nr) v = tr ? s[(long)n * Kr + k] : s[(long)k * Nr + n];
  d[i] = (_Float16)v;
}

__global__ void k_fill(float* p, float v, long n)
{
  long i = (long)blockIdx.x * blockDim.x + threadIdx.x;
  if (i < n) p[i] = v;
}

// f32 [M x Kr] -> f16 [M x Kp] zero-padded (K padded to multiple of 32)
__global__ void k_cvt_pad(const float* __restrict__ s, _Float16* __restrict__ d,
                          long M, int Kr, int Kp)
{
  long i = (long)blockIdx.x * blockDim.x + threadIdx.x;
  if (i >= M * Kp) return;
  long n = i / Kp; int k = (int)(i % Kp);
  d[i] = (k < Kr) ? (_Float16)s[n * Kr + k] : (_Float16)0.f;
}

// wdad[k][h] = sum_c Wd[k][h*64+c] * ad[h][c]   (folds dst-side GAT attention)
__global__ void k_fold_ad(const float* __restrict__ Wd,
                          const float* __restrict__ ad, float* __restrict__ o)
{
  int i = blockIdx.x * blockDim.x + threadIdx.x;
  if (i >= NS * NH) return;
  int k = i / NH, h = i % NH;
  float acc = 0.f;
#pragma unroll 8
  for (int c = 0; c < NS; ++c) acc += Wd[k * NHC + h * NS + c] * ad[h * NS + c];
  o[k * NH + h] = acc;
}

// al_s[n][h] = sum_c hs[n][h*64+c] * a_s[h][c]
__global__ void k_als(const _Float16* __restrict__ hs, const float* __restrict__ av,
                      float* __restrict__ al, long M)
{
  long i = (long)blockIdx.x * blockDim.x + threadIdx.x;
  if (i >= M * NH) return;
  long n = i / NH; int h = (int)(i % NH);
  const _Float16* r = hs + n * NHC + h * NS;
  const float* a = av + h * NS;
  float acc = 0.f;
#pragma unroll 8
  for (int c = 0; c < NS; ++c) acc += (float)r[c] * a[c];
  al[i] = acc;
}

// al_d[n][h] = sum_k x[n][k] * wdad[k][h]
__global__ void k_ald(const _Float16* __restrict__ x, const float* __restrict__ w,
                      float* __restrict__ al, long M)
{
  long i = (long)blockIdx.x * blockDim.x + threadIdx.x;
  if (i >= M * NH) return;
  long n = i / NH; int h = (int)(i % NH);
  const _Float16* r = x + n * NS;
  float acc = 0.f;
#pragma unroll 8
  for (int k = 0; k < NS; ++k) acc += (float)r[k] * w[k * NH + h];
  al[i] = acc;
}

__global__ void k_concat(const _Float16* __restrict__ h1, const _Float16* __restrict__ f1,
                         _Float16* __restrict__ he, long M)
{
  long i = (long)blockIdx.x * blockDim.x + threadIdx.x;
  if (i >= M * 128) return;
  long n = i >> 7; int j = (int)(i & 127);
  he[i] = (j < 64) ? h1[n * 64 + j] : f1[n * 64 + (j - 64)];
}

__device__ __forceinline__ void atomicMaxF(float* a, float v)
{
  if (v >= 0.f) atomicMax((int*)a, __float_as_int(v));
  else          atomicMin((unsigned int*)a, (unsigned int)__float_as_int(v));
}

__device__ __forceinline__ float leaky(float x) { return x > 0.f ? x : 0.2f * x; }

__global__ void k_edge_max(const int* __restrict__ es, const int* __restrict__ ed,
                           const float* __restrict__ als, const float* __restrict__ ald,
                           float* __restrict__ mx, const int* __restrict__ dtype, int tsel)
{
  long i = (long)blockIdx.x * blockDim.x + threadIdx.x;
  if (i >= (long)NE * NH) return;
  long e = i / NH; int h = (int)(i % NH);
  int d = ed[e];
  if (dtype && dtype[d] != tsel) return;
  float l = leaky(als[(long)es[e] * NH + h] + ald[(long)d * NH + h]);
  atomicMaxF(&mx[(long)d * NH + h], l);
}

__global__ void k_edge_exp(const int* __restrict__ es, const int* __restrict__ ed,
                           const float* __restrict__ als, const float* __restrict__ ald,
                           const float* __restrict__ mx, float* __restrict__ ex,
                           float* __restrict__ den, const int* __restrict__ dtype, int tsel)
{
  long i = (long)blockIdx.x * blockDim.x + threadIdx.x;
  if (i >= (long)NE * NH) return;
  long e = i / NH; int h = (int)(i % NH);
  int d = ed[e];
  if (dtype && dtype[d] != tsel) return;
  float l = leaky(als[(long)es[e] * NH + h] + ald[(long)d * NH + h]);
  float v = __expf(l - mx[(long)d * NH + h]);
  ex[i] = v;
  atomicAdd(&den[(long)d * NH + h], v);
}

// one thread per (edge, 4 consecutive channels of one head): E*48 threads
__global__ void k_edge_agg(const int* __restrict__ es, const int* __restrict__ ed,
                           const _Float16* __restrict__ hs, const float* __restrict__ ex,
                           const float* __restrict__ den, float* __restrict__ agg,
                           const int* __restrict__ dtype, int tsel)
{
  long i = (long)blockIdx.x * blockDim.x + threadIdx.x;
  if (i >= (long)NE * 48) return;
  long e = i / 48; int g = (int)(i % 48);
  int d = ed[e];
  if (dtype && dtype[d] != tsel) return;
  int j = g * 4;                                // head-major layout h*64+c
  int h = j >> 6;                               // 4-groups never straddle heads
  float alpha = ex[e * NH + h] / (den[(long)d * NH + h] + 1e-16f);
  const _Float16* hp = hs + (long)es[e] * NHC + j;
  float* ap = agg + (long)d * NHC + j;
#pragma unroll
  for (int u = 0; u < 4; ++u) atomicAdd(ap + u, alpha * (float)hp[u]);
}

// m[n][s] = sum_h (agg[n][s*3+h] + bias[s*3+h])   (torch view(N,64,3).sum(-1))
__global__ void k_pool_post(const float* __restrict__ agg, const float* __restrict__ bias,
                            const int* __restrict__ ftype, _Float16* __restrict__ m, long M)
{
  long i = (long)blockIdx.x * blockDim.x + threadIdx.x;
  if (i >= M * NS) return;
  long n = i >> 6; int s = (int)(i & 63);
  const float* b = bias + (ftype ? (long)ftype[n] * NHC : 0);
  float acc = 0.f;
#pragma unroll
  for (int h = 0; h < NH; ++h) acc += agg[n * NHC + s * NH + h] + b[s * NH + h];
  m[i] = (_Float16)acc;
}

__global__ void k_gru(const _Float16* __restrict__ gx, const _Float16* __restrict__ gh,
                      const _Float16* __restrict__ hp, _Float16* __restrict__ out, long M)
{
  long i = (long)blockIdx.x * blockDim.x + threadIdx.x;
  if (i >= M * NS) return;
  long n = i >> 6; int s = (int)(i & 63);
  float xr = gx[n * NHC + s], xz = gx[n * NHC + 64 + s], xn = gx[n * NHC + 128 + s];
  float hr = gh[n * NHC + s], hz = gh[n * NHC + 64 + s], hn = gh[n * NHC + 128 + s];
  float r = 1.f / (1.f + __expf(-(xr + hr)));
  float z = 1.f / (1.f + __expf(-(xz + hz)));
  float nn = tanhf(xn + r * hn);
  float h0 = (float)hp[i];
  out[i] = (_Float16)((1.f - z) * nn + z * h0);
}

__global__ void k_sel_h(const _Float16* __restrict__ cand, const int* __restrict__ ftype,
                        _Float16* __restrict__ out, long M, int W, long tstride)
{
  long i = (long)blockIdx.x * blockDim.x + threadIdx.x;
  if (i >= M * W) return;
  long n = i / W;
  out[i] = cand[(long)ftype[n] * tstride + i];
}

__global__ void k_sel_f(const float* __restrict__ cand, const int* __restrict__ ftype,
                        float* __restrict__ out, long M, int W, long tstride)
{
  long i = (long)blockIdx.x * blockDim.x + threadIdx.x;
  if (i >= M * W) return;
  long n = i / W;
  out[i] = cand[(long)ftype[n] * tstride + i];
}

static inline long cdiv(long a, long b) { return (a + b - 1) / b; }

extern "C" void kernel_launch(void* const* d_in, const int* in_sizes, int n_in,
                              void* d_out, int out_size, void* d_ws, size_t ws_size,
                              hipStream_t stream)
{
  (void)in_sizes; (void)n_in; (void)out_size; (void)ws_size;
  // ---- inputs (setup_inputs dict order, params flattened) ----
  const float* x1    = (const float*)d_in[0];
  const float* x2    = (const float*)d_in[1];
  const int*   ftype = (const int*)d_in[2];
  const int*   esrc  = (const int*)d_in[3];   // factor indices [0,N1)
  const int*   edst  = (const int*)d_in[4];   // variable indices [0,N2)
  const float* e1W1 = (const float*)d_in[5],  *e1b1 = (const float*)d_in[6];
  const float* e1W2 = (const float*)d_in[7],  *e1b2 = (const float*)d_in[8];
  const float* e1W3 = (const float*)d_in[9],  *e1b3 = (const float*)d_in[10];
  const float* e2W1 = (const float*)d_in[11], *e2b1 = (const float*)d_in[12];
  const float* e2W2 = (const float*)d_in[13], *e2b2 = (const float*)d_in[14];
  const float* e2W3 = (const float*)d_in[15], *e2b3 = (const float*)d_in[16];
  const float* d1W1 = (const float*)d_in[17], *d1b1 = (const float*)d_in[18];
  const float* d1W2 = (const float*)d_in[19], *d1b2 = (const float*)d_in[20];
  const float* d1W3 = (const float*)d_in[21], *d1b3 = (const float*)d_in[22];
  const float* d2W1 = (const float*)d_in[23], *d2b1 = (const float*)d_in[24];
  const float* d2W2 = (const float*)d_in[25], *d2b2 = (const float*)d_in[26];
  const float* d2W3 = (const float*)d_in[27], *d2b3 = (const float*)d_in[28];
  const float* p12Ws = (const float*)d_in[29], *p12Wd = (const float*)d_in[30];
  const float* p12as = (const float*)d_in[31], *p12ad = (const float*)d_in[32];
  const float* p12b  = (const float*)d_in[33];
  const float* p21Ws = (const float*)d_in[34], *p21Wd = (const float*)d_in[35];
  const float* p21as = (const float*)d_in[36], *p21ad = (const float*)d_in[37];
  const float* p21b  = (const float*)d_in[38];
  const float* r1Wih = (const float*)d_in[39], *r1Whh = (const float*)d_in[40];
  const float* r1bih = (const float*)d_in[41], *r1bhh = (const float*)d_in[42];
  const float* r2Wih = (const float*)d_in[43], *r2Whh = (const float*)d_in[44];
  const float* r2bih = (const float*)d_in[45], *r2bhh = (const float*)d_in[46];
  float* out = (float*)d_out;

  // ---- workspace bump allocator ----
  char* base = (char*)d_ws; size_t off = 0;
  auto alloc = [&](size_t bytes) -> char* {
    off = (off + 255) & ~(size_t)255; char* p = base + off; off += bytes; return p;
  };
  auto ah = [&](long n) -> _Float16* { return (_Float16*)alloc((size_t)n * sizeof(_Float16)); };
  auto af = [&](long n) -> float*    { return (float*)alloc((size_t)n * sizeof(float)); };

  // packed f16 weights (B-fragment layout)
  _Float16* wE1W1 = ah(3 * 2048),  *wE1W2 = ah(3 * 4096), *wE1W3 = ah(3 * 4096);
  _Float16* wE2W1 = ah(2048),      *wE2W2 = ah(4096),     *wE2W3 = ah(4096);
  _Float16* wD1W1 = ah(3 * 4096),  *wD1W2 = ah(3 * 4096), *wD1W3 = ah(3 * 1024);
  _Float16* wD2W1 = ah(4096),      *wD2W2 = ah(4096),     *wD2W3 = ah(1024);
  _Float16* wP12Ws = ah(24576),    *wP21Ws = ah(3 * 12288);
  _Float16* wR1Wih = ah(3 * 12288), *wR1Whh = ah(3 * 12288);
  _Float16* wR2Wih = ah(12288),    *wR2Whh = ah(12288);
  float* p12wdad = af(NS * NH);
  float* p21wdad = af(NT * NS * NH);

  // node / edge buffers
  _Float16* x1h  = ah((long)NN1 * 32);
  _Float16* x2h  = ah((long)NN2 * 32);
  _Float16* f1h  = ah((long)NN1 * NS);
  _Float16* h1A  = ah((long)NN1 * NS), *h1B = ah((long)NN1 * NS);
  _Float16* h2A  = ah((long)NN2 * NS), *h2B = ah((long)NN2 * NS);
  _Float16* h1e  = ah((long)NN1 * 128);
  _Float16* tA   = ah((long)NN1 * NS), *tB = ah((long)NN1 * NS);
  _Float16* perT = ah((long)NT * NN1 * NS);
  _Float16* hs1  = ah((long)NN1 * NHC);
  _Float16* hs2  = ah((long)NN2 * NHC);
  _Float16* gxh  = ah((long)NN1 * NHC), *ghh = ah((long)NN1 * NHC);
  _Float16* m1h  = ah((long)NN1 * NS),  *m2h = ah((long)NN2 * NS);
  float* als1 = af((long)NN1 * NH), *ald2 = af((long)NN2 * NH);
  float* als2 = af((long)NN2 * NH), *ald1 = af((long)NN1 * NH);
  float* mx2 = af((long)NN2 * NH), *den2 = af((long)NN2 * NH);
  float* mx1 = af((long)NN1 * NH), *den1 = af((long)NN1 * NH);
  float* exb = af((long)NE * NH);
  float* agg = af((long)NN1 * NHC);        // shared agg arena (covers N2 too)
  float* outT = af((long)NT * NN1 * 4);

  // ---- helpers ----
  auto GEMM = [&](const _Float16* A, const _Float16* Bp, const float* bias,
                  _Float16* Ch, float* Cf, long M, int K, int Np, int ncols, int act) {
    if ((Np & 63) == 0) {
      dim3 g(Np / 64, (unsigned)(M / 16));
      k_gemm<4><<<g, 32, 0, stream>>>(A, Bp, bias, Ch, Cf, K, ncols, act);
    } else {
      dim3 g(Np / 16, (unsigned)(M / 16));
      k_gemm<1><<<g, 32, 0, stream>>>(A, Bp, bias, Ch, Cf, K, ncols, act);
    }
  };
  auto PACK = [&](const float* s, _Float16* dst, int Kr, int Nr, int Kp, int Np, int tr) {
    long tot = (long)(Np >> 4) * (Kp >> 5) * 512;
    k_pack<<<cdiv(tot, 256), 256, 0, stream>>>(s, dst, Kr, Nr, Kp, Np, tr);
  };
  auto FILL = [&](float* p, float v, long n) {
    k_fill<<<cdiv(n, 256), 256, 0, stream>>>(p, v, n);
  };

  // ---- weight packing ----
  for (int t = 0; t < NT; ++t) {
    PACK(e1W1 + (long)t * 8 * 64,   wE1W1 + (long)t * 2048, 8, 64, 32, 64, 0);
    PACK(e1W2 + (long)t * 64 * 64,  wE1W2 + (long)t * 4096, 64, 64, 64, 64, 0);
    PACK(e1W3 + (long)t * 64 * 64,  wE1W3 + (long)t * 4096, 64, 64, 64, 64, 0);
    PACK(d1W1 + (long)t * 64 * 64,  wD1W1 + (long)t * 4096, 64, 64, 64, 64, 0);
    PACK(d1W2 + (long)t * 64 * 64,  wD1W2 + (long)t * 4096, 64, 64, 64, 64, 0);
    PACK(d1W3 + (long)t * 64 * 4,   wD1W3 + (long)t * 1024, 64, 4, 64, 16, 0);
    PACK(p21Ws + (long)t * 64 * NHC, wP21Ws + (long)t * 12288, 64, NHC, 64, NHC, 0);
    PACK(r1Wih + (long)t * NHC * 64, wR1Wih + (long)t * 12288, 64, NHC, 64, NHC, 1);
    PACK(r1Whh + (long)t * NHC * 64, wR1Whh + (long)t * 12288, 64, NHC, 64, NHC, 1);
    k_fold_ad<<<1, 256, 0, stream>>>(p21Wd + (long)t * 64 * NHC,
                                     p21ad + (long)t * NH * NS,
                                     p21wdad + (long)t * NS * NH);
  }
  PACK(e2W1, wE2W1, 8, 64, 32, 64, 0);
  PACK(e2W2, wE2W2, 64, 64, 64, 64, 0);
  PACK(e2W3, wE2W3, 64, 64, 64, 64, 0);
  PACK(d2W1, wD2W1, 64, 64, 64, 64, 0);
  PACK(d2W2, wD2W2, 64, 64, 64, 64, 0);
  PACK(d2W3, wD2W3, 64, 4, 64, 16, 0);
  PACK(p12Ws, wP12Ws, 128, NHC, 128, NHC, 0);
  PACK(r2Wih, wR2Wih, 64, NHC, 64, NHC, 1);
  PACK(r2Whh, wR2Whh, 64, NHC, 64, NHC, 1);
  k_fold_ad<<<1, 256, 0, stream>>>(p12Wd, p12ad, p12wdad);

  // ---- input conversion (pad K=8 -> 32) ----
  k_cvt_pad<<<cdiv((long)NN1 * 32, 256), 256, 0, stream>>>(x1, x1h, NN1, 8, 32);
  k_cvt_pad<<<cdiv((long)NN2 * 32, 256), 256, 0, stream>>>(x2, x2h, NN2, 8, 32);

  // ---- encoders ----
  for (int t = 0; t < NT; ++t) {
    GEMM(x1h, wE1W1 + (long)t * 2048, e1b1 + t * 64, tA, nullptr, NN1, 32, 64, 64, 1);
    GEMM(tA, wE1W2 + (long)t * 4096, e1b2 + t * 64, tB, nullptr, NN1, 64, 64, 64, 1);
    GEMM(tB, wE1W3 + (long)t * 4096, e1b3 + t * 64, perT + (long)t * NN1 * NS, nullptr,
         NN1, 64, 64, 64, 0);
  }
  k_sel_h<<<cdiv((long)NN1 * NS, 256), 256, 0, stream>>>(perT, ftype, f1h, NN1, NS,
                                                         (long)NN1 * NS);
  hipMemcpyAsync(h1A, f1h, (size_t)NN1 * NS * sizeof(_Float16),
                 hipMemcpyDeviceToDevice, stream);
  GEMM(x2h, wE2W1, e2b1, tA, nullptr, NN2, 32, 64, 64, 1);
  GEMM(tA, wE2W2, e2b2, tB, nullptr, NN2, 64, 64, 64, 1);
  GEMM(tB, wE2W3, e2b3, h2A, nullptr, NN2, 64, 64, 64, 0);

  const long EH = (long)NE * NH, EC4 = (long)NE * 48;

  // ---- message passing steps ----
  for (int step = 0; step < 2; ++step) {
    // --- m2: factor -> variable (GAT p12) ---
    k_concat<<<cdiv((long)NN1 * 128, 256), 256, 0, stream>>>(h1A, f1h, h1e, NN1);
    GEMM(h1e, wP12Ws, nullptr, hs1, nullptr, NN1, 128, NHC, NHC, 0);
    k_als<<<cdiv((long)NN1 * NH, 256), 256, 0, stream>>>(hs1, p12as, als1, NN1);
    k_ald<<<cdiv((long)NN2 * NH, 256), 256, 0, stream>>>(h2A, p12wdad, ald2, NN2);
    FILL(mx2, -1e30f, (long)NN2 * NH);
    FILL(den2, 0.f, (long)NN2 * NH);
    FILL(agg, 0.f, (long)NN2 * NHC);
    k_edge_max<<<cdiv(EH, 256), 256, 0, stream>>>(esrc, edst, als1, ald2, mx2, nullptr, 0);
    k_edge_exp<<<cdiv(EH, 256), 256, 0, stream>>>(esrc, edst, als1, ald2, mx2, exb, den2,
                                                  nullptr, 0);
    k_edge_agg<<<cdiv(EC4, 256), 256, 0, stream>>>(esrc, edst, hs1, exb, den2, agg,
                                                   nullptr, 0);
    k_pool_post<<<cdiv((long)NN2 * NS, 256), 256, 0, stream>>>(agg, p12b, nullptr, m2h, NN2);

    // --- m1: variable -> factor (per-type GAT p21, masked edge passes) ---
    FILL(mx1, -1e30f, (long)NN1 * NH);
    FILL(den1, 0.f, (long)NN1 * NH);
    FILL(agg, 0.f, (long)NN1 * NHC);
    for (int t = 0; t < NT; ++t) {
      GEMM(h2A, wP21Ws + (long)t * 12288, nullptr, hs2, nullptr, NN2, 64, NHC, NHC, 0);
      k_als<<<cdiv((long)NN2 * NH, 256), 256, 0, stream>>>(hs2, p21as + (long)t * NH * NS,
                                                           als2, NN2);
      k_ald<<<cdiv((long)NN1 * NH, 256), 256, 0, stream>>>(h1A, p21wdad + (long)t * NS * NH,
                                                           ald1, NN1);
      k_edge_max<<<cdiv(EH, 256), 256, 0, stream>>>(edst, esrc, als2, ald1, mx1, ftype, t);
      k_edge_exp<<<cdiv(EH, 256), 256, 0, stream>>>(edst, esrc, als2, ald1, mx1, exb, den1,
                                                    ftype, t);
      k_edge_agg<<<cdiv(EC4, 256), 256, 0, stream>>>(edst, esrc, hs2, exb, den1, agg,
                                                     ftype, t);
    }
    k_pool_post<<<cdiv((long)NN1 * NS, 256), 256, 0, stream>>>(agg, p21b, ftype, m1h, NN1);

    // --- GRU updates ---
    GEMM(m2h, wR2Wih, r2bih, gxh, nullptr, NN2, 64, NHC, NHC, 0);
    GEMM(h2A, wR2Whh, r2bhh, ghh, nullptr, NN2, 64, NHC, NHC, 0);
    k_gru<<<cdiv((long)NN2 * NS, 256), 256, 0, stream>>>(gxh, ghh, h2A, h2B, NN2);
    for (int t = 0; t < NT; ++t) {
      GEMM(m1h, wR1Wih + (long)t * 12288, r1bih + (long)t * NHC, gxh, nullptr,
           NN1, 64, NHC, NHC, 0);
      GEMM(h1A, wR1Whh + (long)t * 12288, r1bhh + (long)t * NHC, ghh, nullptr,
           NN1, 64, NHC, NHC, 0);
      k_gru<<<cdiv((long)NN1 * NS, 256), 256, 0, stream>>>(gxh, ghh, h1A,
                                                           perT + (long)t * NN1 * NS, NN1);
    }
    k_sel_h<<<cdiv((long)NN1 * NS, 256), 256, 0, stream>>>(perT, ftype, h1B, NN1, NS,
                                                           (long)NN1 * NS);
    _Float16* tmp;
    tmp = h1A; h1A = h1B; h1B = tmp;
    tmp = h2A; h2A = h2B; h2B = tmp;
  }

  // ---- decoders ----
  for (int t = 0; t < NT; ++t) {
    GEMM(h1A, wD1W1 + (long)t * 4096, d1b1 + t * 64, tA, nullptr, NN1, 64, 64, 64, 1);
    GEMM(tA, wD1W2 + (long)t * 4096, d1b2 + t * 64, tB, nullptr, NN1, 64, 64, 64, 1);
    GEMM(tB, wD1W3 + (long)t * 1024, d1b3 + t * 4, nullptr, outT + (long)t * NN1 * 4,
         NN1, 64, 16, 4, 0);
  }
  k_sel_f<<<cdiv((long)NN1 * 4, 256), 256, 0, stream>>>(outT, ftype, out, NN1, 4,
                                                        (long)NN1 * 4);
  GEMM(h2A, wD2W1, d2b1, tA, nullptr, NN2, 64, 64, 64, 1);
  GEMM(tA, wD2W2, d2b2, tB, nullptr, NN2, 64, 64, 64, 1);
  GEMM(tB, wD2W3, d2b3, nullptr, out + (long)NN1 * 4, NN2, 64, 16, 4, 0);
}
